// Model_13391708029601
// MI455X (gfx1250) — matrix operand order repaired and checked by reading, not compiled, not run on hardware
//
#include <hip/hip_runtime.h>
#include <math.h>

// ---------------------------------------------------------------------------
// Geo-FNO forward for MI455X (gfx1250, wave32, WMMA).
// Heavy NUFFT projections run on v_wmma_f32_16x16x32_f16 (f32 accumulate).
// LDS tiles are staged in WMMA *fragment order* so each operand is a single
// 32-byte LDS vector load (2x ds_load_b128) instead of 16x ds_load_u16.
// ---------------------------------------------------------------------------

#define G      20000
#define WDTH   64
#define S      96
#define MODES  12
#define M2     24      // 2*MODES
#define M23    23      // 2*MODES-1
#define NMODE  552     // 24*23
#define NMODEP 576     // padded to multiple of 32
#define KCHUNK 25      // project_in K split: 25 chunks of 800 (=25 tiles of 32)

#define PI_F      3.14159265358979323846f
#define TWO_PI_F  6.28318530717958647692f

typedef __attribute__((ext_vector_type(16))) _Float16 v16h;
typedef __attribute__((ext_vector_type(8)))  _Float16 v8h;
typedef __attribute__((ext_vector_type(8)))  float    v8f;

__device__ __forceinline__ float gelu_exact(float v) {
    return 0.5f * v * (1.0f + erff(v * 0.70710678118654752440f));
}

// A-matrix (16-bit, 16x32) K index for half h (0..15) in lane (0..31).
// ISA 7.12.2: lanes 0-15: VGPR0..3 -> K=0..7, VGPR4..7 -> K=16..23;
//             lanes 16-31: same +8.
__device__ __forceinline__ int a_kmap(int h, int lane) {
    int v = h >> 1;
    return 2 * (v & 3) + (h & 1) + ((v >> 2) << 4) + ((lane >> 4) << 3);
}

// B-matrix (16-bit, 32x16) K index for half h in lane: same order as A (K = (h&7) + 16*(h>>3) + 8*(lane>>4)).
__device__ __forceinline__ int b_kmap(int h, int lane) {
    return (h & 7) + ((h >> 3) << 4) + ((lane >> 4) << 3);
}

// ---------------------------------------------------------------------------
// Stage 1: pre-MLP  (b,g,5) -> GELU -> (b,g,128) -> (b,g,64); store as u[c][n]
// ---------------------------------------------------------------------------
__global__ void k_pre(const float* __restrict__ x, const float* __restrict__ coords,
                      const float* __restrict__ w1, const float* __restrict__ b1,
                      const float* __restrict__ w2, const float* __restrict__ b2,
                      float* __restrict__ u) {
    int n = blockIdx.x * blockDim.x + threadIdx.x;
    if (n >= G) return;
    float in5[5];
    in5[0] = x[0 * G + n]; in5[1] = x[1 * G + n]; in5[2] = x[2 * G + n];
    in5[3] = coords[0 * G + n]; in5[4] = coords[1 * G + n];
    float t[128];
    for (int j = 0; j < 128; ++j) {
        float s = b1[j];
        for (int c = 0; c < 5; ++c) s += in5[c] * w1[c * 128 + j];
        t[j] = gelu_exact(s);
    }
    for (int c = 0; c < WDTH; ++c) {
        float s = b2[c];
        for (int j = 0; j < 128; ++j) s += t[j] * w2[j * WDTH + c];
        u[c * G + n] = s;
    }
}

// ---------------------------------------------------------------------------
// Stage 2: IPHI coordinate warp. One block (128 threads) per point.
// ---------------------------------------------------------------------------
__global__ void k_iphi(const float* __restrict__ coords,
                       const float* __restrict__ fc0w, const float* __restrict__ fc0b,
                       const float* __restrict__ fncw, const float* __restrict__ fncb,
                       const float* __restrict__ f1w,  const float* __restrict__ f1b,
                       const float* __restrict__ f2w,  const float* __restrict__ f2b,
                       const float* __restrict__ f3w,  const float* __restrict__ f3b,
                       const float* __restrict__ f4w,  const float* __restrict__ f4b,
                       float* __restrict__ xphi) {
    __shared__ float cat[96];
    __shared__ float bufA[128];
    __shared__ float bufB[128];
    int n = blockIdx.x;
    int t = threadIdx.x;
    float p0 = coords[n], p1 = coords[G + n];
    float xc0 = p0 - 1e-4f, xc1 = p1 - 1e-4f;
    float xd[4];
    xd[0] = p0; xd[1] = p1;
    xd[2] = atan2f(xc1, xc0);
    xd[3] = sqrtf(xc0 * xc0 + xc1 * xc1);
    if (t < 32) {
        float s = fc0b[t];
        for (int d = 0; d < 4; ++d) s += xd[d] * fc0w[d * 32 + t];
        cat[t] = s;
    } else if (t < 96) {
        int idx = t - 32;            // 0..63
        int which = idx >> 5;        // 0 = sin block, 1 = cos block
        int q = idx & 31;
        int dim = q >> 3, j = q & 7;
        float f = xd[dim] * (PI_F * (float)(1 << j));
        cat[t] = which ? cosf(f) : sinf(f);
    }
    __syncthreads();
    { float s = fncb[t]; for (int k = 0; k < 96;  ++k) s += cat[k]  * fncw[k * 128 + t]; bufA[t] = s; }
    __syncthreads();
    { float s = f1b[t];  for (int k = 0; k < 128; ++k) s += bufA[k] * f1w[k * 128 + t];  bufB[t] = tanhf(s); }
    __syncthreads();
    { float s = f2b[t];  for (int k = 0; k < 128; ++k) s += bufB[k] * f2w[k * 128 + t];  bufA[t] = tanhf(s); }
    __syncthreads();
    { float s = f3b[t];  for (int k = 0; k < 128; ++k) s += bufA[k] * f3w[k * 128 + t];  bufB[t] = tanhf(s); }
    __syncthreads();
    if (t < 2) {
        float s = f4b[t];
        for (int k = 0; k < 128; ++k) s += bufB[k] * f4w[k * 2 + t];
        float p = (t == 0) ? p0 : p1;
        xphi[t * G + n] = p + p * s;
    }
}

// ---------------------------------------------------------------------------
// Stage 3: separable basis phase tables e1 = exp(-2pi i x0 k1), e2 likewise.
// ---------------------------------------------------------------------------
__global__ void k_basis(const float* __restrict__ xphi,
                        float* __restrict__ e1r, float* __restrict__ e1i,
                        float* __restrict__ e2r, float* __restrict__ e2i) {
    int n = blockIdx.x * blockDim.x + threadIdx.x;
    if (n >= G) return;
    float x0 = xphi[n], x1 = xphi[G + n];
    for (int r = 0; r < M2; ++r) {
        float k1 = (float)(r < MODES ? r : r - M2);
        float s, c; sincosf(-TWO_PI_F * x0 * k1, &s, &c);
        e1r[r * G + n] = c; e1i[r * G + n] = s;
    }
    for (int q = 0; q < M23; ++q) {
        float k2 = (float)(q < MODES ? q : q - M23);
        float s, c; sincosf(-TWO_PI_F * x1 * k2, &s, &c);
        e2r[q * G + n] = c; e2i[q * G + n] = s;
    }
}

// ---------------------------------------------------------------------------
// Stage 4: project_in GEMM:  uft[c,kx,ky] = sum_n u[c,n] * e1[kx,n]*e2[ky,n]
// WMMA f16 -> f32, per-chunk partials (deterministic, no atomics).
// LDS staged in fragment order: AsF[mt][lane][h], BsF[part][lane][h].
// partial layout: [(chunk*2+part)][c][kx][ky<12]
// ---------------------------------------------------------------------------
__global__ void __launch_bounds__(256)
k_pin_gemm(const float* __restrict__ u,
           const float* __restrict__ e1r, const float* __restrict__ e1i,
           const float* __restrict__ e2r, const float* __restrict__ e2i,
           float* __restrict__ partial) {
    __shared__ __align__(32) _Float16 AsF[4][32][16];   // 4 KB
    __shared__ __align__(32) _Float16 BsF[2][32][16];   // 2 KB
    int kx = blockIdx.x;
    int chunk = blockIdx.y;
    int tid = threadIdx.x;
    int lane = tid & 31, wid = tid >> 5;
    int mt = wid & 3, part = wid >> 2;   // part: 0 = re, 1 = im
    v8f acc = {};
    int nbase = chunk * (G / KCHUNK);    // 800 points per chunk
    for (int kt = 0; kt < 25; ++kt) {
        int n0 = nbase + kt * 32;
        // A loader: 2048 halves, 8 per thread, written in fragment order.
        {
            int idx0 = tid * 8;
            int amt = idx0 >> 9, alane = (idx0 >> 4) & 31, h0 = idx0 & 15;
            int c = amt * 16 + (alane & 15);
            const float* row = u + (size_t)c * G + n0;
            v8h vals;
#pragma unroll
            for (int e = 0; e < 8; ++e)
                vals[e] = (_Float16)row[a_kmap(h0 + e, alane)];
            *(v8h*)(&AsF[amt][alane][h0]) = vals;
            __builtin_prefetch(row + 32, 0, 1);   // next K tile (global_prefetch_b8)
        }
        // B loader: 512 (K,N) products, 2 per thread, re+im written together.
        {
            int idx0 = tid * 2;
            int blane = (idx0 >> 4) & 31, h0 = idx0 & 15;
            int col = blane & 15;
#pragma unroll
            for (int e = 0; e < 2; ++e) {
                int h = h0 + e;
                int kk = b_kmap(h, blane);
                float pr = 0.f, pi = 0.f;
                if (col < MODES) {
                    int n = n0 + kk;
                    float ar = e1r[kx * G + n], ai = e1i[kx * G + n];
                    float br = e2r[col * G + n], bi = e2i[col * G + n];
                    pr = ar * br - ai * bi;
                    pi = ar * bi + ai * br;
                }
                BsF[0][blane][h] = (_Float16)pr;
                BsF[1][blane][h] = (_Float16)pi;
            }
        }
        __syncthreads();
        v16h af = *(const v16h*)(&AsF[mt][lane][0]);
        v16h bf = *(const v16h*)(&BsF[part][lane][0]);
        acc = __builtin_amdgcn_wmma_f32_16x16x32_f16(false, af, false, bf,
                                                     (short)0, acc, false, false);
        __syncthreads();
    }
    int N = lane & 15;
    if (N < MODES) {
        for (int r = 0; r < 8; ++r) {
            int c = mt * 16 + r + ((lane >> 4) << 3);
            partial[(size_t)(chunk * 2 + part) * (64 * M2 * MODES)
                    + (c * M2 + kx) * MODES + N] = acc[r];
        }
    }
}

__global__ void k_reduce_pin(const float* __restrict__ partial,
                             float* __restrict__ uftr, float* __restrict__ ufti) {
    int idx = blockIdx.x * blockDim.x + threadIdx.x;   // 2 * 64*24*12
    const int NN = 64 * M2 * MODES;
    if (idx >= 2 * NN) return;
    int part = idx / NN, j = idx % NN;
    float s = 0.f;
    for (int ch = 0; ch < KCHUNK; ++ch) s += partial[(size_t)(ch * 2 + part) * NN + j];
    (part ? ufti : uftr)[j] = s;
}

// ---------------------------------------------------------------------------
// Spectral channel mix (shared by project_in and project_out):
// m[o,mrow,ky] = sum_i uft[i,mrow,ky] * w{1|2}[i,o,x,ky]   (complex)
// ---------------------------------------------------------------------------
__global__ void k_specmix(const float* __restrict__ ur, const float* __restrict__ ui,
                          const float* __restrict__ w1r, const float* __restrict__ w1i,
                          const float* __restrict__ w2r, const float* __restrict__ w2i,
                          float* __restrict__ mr, float* __restrict__ mi) {
    int idx = blockIdx.x * blockDim.x + threadIdx.x;   // 64*24*12
    if (idx >= 64 * M2 * MODES) return;
    int ky = idx % MODES, m = (idx / MODES) % M2, o = idx / (M2 * MODES);
    const float *wr, *wi;
    int xw = (m < MODES) ? m : m - MODES;
    if (m < MODES) { wr = w1r; wi = w1i; } else { wr = w2r; wi = w2i; }
    float sr = 0.f, si = 0.f;
    for (int i = 0; i < 64; ++i) {
        float a = ur[(i * M2 + m) * MODES + ky];
        float b = ui[(i * M2 + m) * MODES + ky];
        int widx = ((i * 64 + o) * MODES + xw) * MODES + ky;
        float c = wr[widx], d = wi[widx];
        sr += a * c - b * d;
        si += a * d + b * c;
    }
    mr[idx] = sr; mi[idx] = si;
}

// ---------------------------------------------------------------------------
// irfft2 of sparse spectrum (24 row-freqs x 12 col-freqs), two stages.
// ---------------------------------------------------------------------------
__global__ void k_grid1(const float* __restrict__ mr, const float* __restrict__ mi,
                        float* __restrict__ gr, float* __restrict__ gi) {
    int idx = blockIdx.x * blockDim.x + threadIdx.x;   // 64*96*12
    if (idx >= 64 * S * MODES) return;
    int ky = idx % MODES, a = (idx / MODES) % S, o = idx / (S * MODES);
    float sr = 0.f, si = 0.f;
    for (int m = 0; m < M2; ++m) {
        int k = (m < MODES) ? m : m - M2;
        float s, c; sincosf(TWO_PI_F * (float)(k * a) / (float)S, &s, &c);
        float ar = mr[o * (M2 * MODES) + m * MODES + ky];
        float ai = mi[o * (M2 * MODES) + m * MODES + ky];
        sr += ar * c - ai * s;
        si += ar * s + ai * c;
    }
    gr[idx] = sr; gi[idx] = si;
}

__global__ void k_grid2(const float* __restrict__ gr, const float* __restrict__ gi,
                        float* __restrict__ ug) {
    int idx = blockIdx.x * blockDim.x + threadIdx.x;   // 64*96*96
    if (idx >= 64 * S * S) return;
    int b = idx % S, a = (idx / S) % S, o = idx / (S * S);
    int base = o * (S * MODES) + a * MODES;
    float acc = gr[base + 0];
    for (int ky = 1; ky < MODES; ++ky) {
        float s, c; sincosf(TWO_PI_F * (float)(ky * b) / (float)S, &s, &c);
        acc += 2.f * (gr[base + ky] * c - gi[base + ky] * s);
    }
    ug[idx] = acc * (1.f / (float)(S * S));
}

// ---------------------------------------------------------------------------
// F-FNO block pieces (direct 12-mode DFTs + complex channel mix).
// ---------------------------------------------------------------------------
__global__ void k_ffty(const float* __restrict__ u, float* __restrict__ fr, float* __restrict__ fi) {
    int idx = blockIdx.x * blockDim.x + threadIdx.x;   // 64*96*12 : [c][x][ky]
    if (idx >= 64 * S * MODES) return;
    int ky = idx % MODES, x = (idx / MODES) % S, c = idx / (S * MODES);
    float sr = 0.f, si = 0.f;
    for (int b = 0; b < S; ++b) {
        float s, cs; sincosf(TWO_PI_F * (float)(ky * b) / (float)S, &s, &cs);
        float v = u[c * (S * S) + x * S + b];
        sr += v * cs; si -= v * s;
    }
    fr[idx] = sr; fi[idx] = si;
}

__global__ void k_fftx(const float* __restrict__ u, float* __restrict__ fr, float* __restrict__ fi) {
    int idx = blockIdx.x * blockDim.x + threadIdx.x;   // 64*12*96 : [c][kx][y]
    if (idx >= 64 * MODES * S) return;
    int y = idx % S, kx = (idx / S) % MODES, c = idx / (S * MODES);
    float sr = 0.f, si = 0.f;
    for (int a = 0; a < S; ++a) {
        float s, cs; sincosf(TWO_PI_F * (float)(kx * a) / (float)S, &s, &cs);
        float v = u[c * (S * S) + a * S + y];
        sr += v * cs; si -= v * s;
    }
    fr[idx] = sr; fi[idx] = si;
}

__global__ void k_mixy(const float* __restrict__ fr, const float* __restrict__ fi,
                       const float* __restrict__ wr, const float* __restrict__ wi,
                       float* __restrict__ orr, float* __restrict__ oi) {
    int idx = blockIdx.x * blockDim.x + threadIdx.x;   // [o][x][ky]
    if (idx >= 64 * S * MODES) return;
    int ky = idx % MODES, x = (idx / MODES) % S, o = idx / (S * MODES);
    float sr = 0.f, si = 0.f;
    for (int i = 0; i < 64; ++i) {
        float a = fr[i * (S * MODES) + x * MODES + ky];
        float b = fi[i * (S * MODES) + x * MODES + ky];
        int widx = i * (64 * MODES) + o * MODES + ky;
        float c = wr[widx], d = wi[widx];
        sr += a * c - b * d;
        si += a * d + b * c;
    }
    orr[idx] = sr; oi[idx] = si;
}

__global__ void k_mixx(const float* __restrict__ fr, const float* __restrict__ fi,
                       const float* __restrict__ wr, const float* __restrict__ wi,
                       float* __restrict__ orr, float* __restrict__ oi) {
    int idx = blockIdx.x * blockDim.x + threadIdx.x;   // [o][kx][y]
    if (idx >= 64 * MODES * S) return;
    int y = idx % S, kx = (idx / S) % MODES, o = idx / (S * MODES);
    float sr = 0.f, si = 0.f;
    for (int i = 0; i < 64; ++i) {
        float a = fr[i * (S * MODES) + kx * S + y];
        float b = fi[i * (S * MODES) + kx * S + y];
        int widx = i * (64 * MODES) + o * MODES + kx;
        float c = wr[widx], d = wi[widx];
        sr += a * c - b * d;
        si += a * d + b * c;
    }
    orr[idx] = sr; oi[idx] = si;
}

__global__ void k_combine(const float* __restrict__ src,
                          const float* __restrict__ oxr, const float* __restrict__ oxi,
                          const float* __restrict__ oyr, const float* __restrict__ oyi,
                          float* __restrict__ dst) {
    int idx = blockIdx.x * blockDim.x + threadIdx.x;   // 64*96*96
    if (idx >= 64 * S * S) return;
    int b = idx % S, a = (idx / S) % S, o = idx / (S * S);
    int ybase = o * (S * MODES) + a * MODES;
    float uy = oyr[ybase + 0];
    for (int ky = 1; ky < MODES; ++ky) {
        float s, c; sincosf(TWO_PI_F * (float)(ky * b) / (float)S, &s, &c);
        uy += 2.f * (oyr[ybase + ky] * c - oyi[ybase + ky] * s);
    }
    int xbase = o * (S * MODES) + b;
    float ux = oxr[xbase + 0];
    for (int kx = 1; kx < MODES; ++kx) {
        float s, c; sincosf(TWO_PI_F * (float)(kx * a) / (float)S, &s, &c);
        ux += 2.f * (oxr[xbase + kx * S] * c - oxi[xbase + kx * S] * s);
    }
    dst[idx] = src[idx] + (ux + uy) * (1.f / (float)S);
}

// ---------------------------------------------------------------------------
// project_out: forward rfft2 (24 row-freqs, 12 col-freqs), two stages.
// ---------------------------------------------------------------------------
__global__ void k_po1(const float* __restrict__ u, float* __restrict__ tr, float* __restrict__ ti) {
    int idx = blockIdx.x * blockDim.x + threadIdx.x;   // 64*24*96 : [c][m][b]
    if (idx >= 64 * M2 * S) return;
    int b = idx % S, m = (idx / S) % M2, c = idx / (M2 * S);
    int k = (m < MODES) ? m : m - M2;
    float sr = 0.f, si = 0.f;
    for (int a = 0; a < S; ++a) {
        float s, cs; sincosf(TWO_PI_F * (float)(k * a) / (float)S, &s, &cs);
        float v = u[c * (S * S) + a * S + b];
        sr += v * cs; si -= v * s;
    }
    tr[idx] = sr; ti[idx] = si;
}

__global__ void k_po2(const float* __restrict__ tr, const float* __restrict__ ti,
                      float* __restrict__ u2r, float* __restrict__ u2i) {
    int idx = blockIdx.x * blockDim.x + threadIdx.x;   // 64*24*12 : [c][m][ky]
    if (idx >= 64 * M2 * MODES) return;
    int ky = idx % MODES, m = (idx / MODES) % M2, c = idx / (M2 * MODES);
    float sr = 0.f, si = 0.f;
    for (int b = 0; b < S; ++b) {
        float s, cs; sincosf(TWO_PI_F * (float)(ky * b) / (float)S, &s, &cs);
        float ar = tr[c * (M2 * S) + m * S + b];
        float ai = ti[c * (M2 * S) + m * S + b];
        sr += ar * cs + ai * s;          // *(cs - i s)
        si += -ar * s + ai * cs;
    }
    u2r[idx] = sr; u2i[idx] = si;
}

// Hermitian completion + flatten to padded (64 x 576) matrix for WMMA.
__global__ void k_fpack(const float* __restrict__ ftr, const float* __restrict__ fti,
                        float* __restrict__ Fr, float* __restrict__ Fi) {
    int idx = blockIdx.x * blockDim.x + threadIdx.x;   // 64*576
    if (idx >= 64 * NMODEP) return;
    int mo = idx % NMODEP, o = idx / NMODEP;
    float fr = 0.f, fi = 0.f;
    if (mo < NMODE) {
        int r = mo / M23, col = mo % M23;
        if (col < MODES) {
            fr = ftr[o * (M2 * MODES) + r * MODES + col];
            fi = fti[o * (M2 * MODES) + r * MODES + col];
        } else {
            int sr_ = M23 - r, sc = M23 - col;     // conj(out_ft[23-r, 23-col])
            fr =  ftr[o * (M2 * MODES) + sr_ * MODES + sc];
            fi = -fti[o * (M2 * MODES) + sr_ * MODES + sc];
        }
    }
    Fr[idx] = fr; Fi[idx] = fi;
}

// project_out GEMM: uo[c,n] = Re( sum_mo F[c,mo] * conj(e1*e2) )
//                          = sum Fr*pr + Fi*pi  with p = e1*e2 (sign -1 tables)
__global__ void __launch_bounds__(256)
k_pout_gemm(const float* __restrict__ Fr, const float* __restrict__ Fi,
            const float* __restrict__ e1r, const float* __restrict__ e1i,
            const float* __restrict__ e2r, const float* __restrict__ e2i,
            float* __restrict__ uo) {
    __shared__ __align__(32) _Float16 ArF[4][32][16];   // 4 KB
    __shared__ __align__(32) _Float16 AiF[4][32][16];   // 4 KB
    __shared__ __align__(32) _Float16 BrF[2][32][16];   // 2 KB
    __shared__ __align__(32) _Float16 BiF[2][32][16];   // 2 KB
    int tid = threadIdx.x;
    int lane = tid & 31, wid = tid >> 5;
    int mt = wid & 3, nt = wid >> 2;       // nt: 0..1
    int n0 = blockIdx.x * 32;
    v8f acc = {};
    for (int kt = 0; kt < NMODEP / 32; ++kt) {
        int mo0 = kt * 32;
        // A loader: fragment-order Fr and Fi (2048 halves each side; 8+8/thread)
        {
            int idx0 = tid * 8;
            int amt = idx0 >> 9, alane = (idx0 >> 4) & 31, h0 = idx0 & 15;
            int c = amt * 16 + (alane & 15);
            const float* rr = Fr + (size_t)c * NMODEP + mo0;
            const float* ri = Fi + (size_t)c * NMODEP + mo0;
            v8h vr, vi;
#pragma unroll
            for (int e = 0; e < 8; ++e) {
                int K = a_kmap(h0 + e, alane);
                vr[e] = (_Float16)rr[K];
                vi[e] = (_Float16)ri[K];
            }
            *(v8h*)(&ArF[amt][alane][h0]) = vr;
            *(v8h*)(&AiF[amt][alane][h0]) = vi;
            __builtin_prefetch(rr + 32, 0, 1);
            __builtin_prefetch(ri + 32, 0, 1);
        }
        // B loader: 1024 (K,N) basis products, 4 per thread, fragment order.
        {
            int idx0 = tid * 4;
            int bnt = idx0 >> 9, blane = (idx0 >> 4) & 31, h0 = idx0 & 15;
            int n = n0 + bnt * 16 + (blane & 15);
#pragma unroll
            for (int e = 0; e < 4; ++e) {
                int h = h0 + e;
                int mo = mo0 + b_kmap(h, blane);
                float pr = 0.f, pi = 0.f;
                if (mo < NMODE) {
                    int r = mo / M23, col = mo % M23;
                    float ar = e1r[r * G + n], ai2 = e1i[r * G + n];
                    float br = e2r[col * G + n], bi = e2i[col * G + n];
                    pr = ar * br - ai2 * bi;
                    pi = ar * bi + ai2 * br;
                }
                BrF[bnt][blane][h] = (_Float16)pr;
                BiF[bnt][blane][h] = (_Float16)pi;
            }
        }
        __syncthreads();
        v16h a0 = *(const v16h*)(&ArF[mt][lane][0]);
        v16h a1 = *(const v16h*)(&AiF[mt][lane][0]);
        v16h b0 = *(const v16h*)(&BrF[nt][lane][0]);
        v16h b1 = *(const v16h*)(&BiF[nt][lane][0]);
        acc = __builtin_amdgcn_wmma_f32_16x16x32_f16(false, a0, false, b0, (short)0, acc, false, false);
        acc = __builtin_amdgcn_wmma_f32_16x16x32_f16(false, a1, false, b1, (short)0, acc, false, false);
        __syncthreads();
    }
    int N = n0 + nt * 16 + (lane & 15);
    for (int r = 0; r < 8; ++r) {
        int c = mt * 16 + r + ((lane >> 4) << 3);
        uo[c * G + N] = acc[r];
    }
}

// ---------------------------------------------------------------------------
// head MLP: (g,64) -> GELU -> (g,64) -> (g,3)
// ---------------------------------------------------------------------------
__global__ void k_head(const float* __restrict__ uo,
                       const float* __restrict__ w1, const float* __restrict__ b1,
                       const float* __restrict__ w2, const float* __restrict__ b2,
                       float* __restrict__ out) {
    int n = blockIdx.x * blockDim.x + threadIdx.x;
    if (n >= G) return;
    float h[64], t[64];
    for (int c = 0; c < 64; ++c) h[c] = uo[c * G + n];
    for (int j = 0; j < 64; ++j) {
        float s = b1[j];
        for (int c = 0; c < 64; ++c) s += h[c] * w1[c * 64 + j];
        t[j] = gelu_exact(s);
    }
    for (int o = 0; o < 3; ++o) {
        float s = b2[o];
        for (int j = 0; j < 64; ++j) s += t[j] * w2[j * 3 + o];
        out[o * G + n] = s;
    }
}

// ---------------------------------------------------------------------------
extern "C" void kernel_launch(void* const* d_in, const int* in_sizes, int n_in,
                              void* d_out, int out_size, void* d_ws, size_t ws_size,
                              hipStream_t stream) {
    (void)in_sizes; (void)n_in; (void)out_size; (void)ws_size;
    const float* x         = (const float*)d_in[0];
    const float* coords    = (const float*)d_in[1];
    const float* pre_w1    = (const float*)d_in[2];
    const float* pre_b1    = (const float*)d_in[3];
    const float* pre_w2    = (const float*)d_in[4];
    const float* pre_b2    = (const float*)d_in[5];
    const float* pin_w1_re = (const float*)d_in[6];
    const float* pin_w1_im = (const float*)d_in[7];
    const float* pin_w2_re = (const float*)d_in[8];
    const float* pin_w2_im = (const float*)d_in[9];
    const float* pout_w1_re= (const float*)d_in[10];
    const float* pout_w1_im= (const float*)d_in[11];
    const float* pout_w2_re= (const float*)d_in[12];
    const float* pout_w2_im= (const float*)d_in[13];
    const float* ifc0w = (const float*)d_in[14]; const float* ifc0b = (const float*)d_in[15];
    const float* ifncw = (const float*)d_in[16]; const float* ifncb = (const float*)d_in[17];
    const float* ifc1w = (const float*)d_in[18]; const float* ifc1b = (const float*)d_in[19];
    const float* ifc2w = (const float*)d_in[20]; const float* ifc2b = (const float*)d_in[21];
    const float* ifc3w = (const float*)d_in[22]; const float* ifc3b = (const float*)d_in[23];
    const float* ifc4w = (const float*)d_in[24]; const float* ifc4b = (const float*)d_in[25];
    const float* spec_wx_re = (const float*)d_in[26];
    const float* spec_wx_im = (const float*)d_in[27];
    const float* spec_wy_re = (const float*)d_in[28];
    const float* spec_wy_im = (const float*)d_in[29];
    const float* head_w1 = (const float*)d_in[30]; const float* head_b1 = (const float*)d_in[31];
    const float* head_w2 = (const float*)d_in[32]; const float* head_b2 = (const float*)d_in[33];

    char* wsp = (char*)d_ws;
    auto carve = [&](size_t nfloats) { float* p = (float*)wsp; wsp += nfloats * sizeof(float); return p; };

    float* u_pre   = carve((size_t)64 * G);
    float* xphi    = carve((size_t)2 * G);
    float* e1r = carve((size_t)M2 * G);  float* e1i = carve((size_t)M2 * G);
    float* e2r = carve((size_t)M23 * G); float* e2i = carve((size_t)M23 * G);
    float* partial = carve((size_t)KCHUNK * 2 * 64 * M2 * MODES);
    float* uftr = carve(64 * M2 * MODES); float* ufti = carve(64 * M2 * MODES);
    float* mxr  = carve(64 * M2 * MODES); float* mxi  = carve(64 * M2 * MODES);
    float* g1r  = carve(64 * S * MODES);  float* g1i  = carve(64 * S * MODES);
    float* ua   = carve((size_t)64 * S * S);
    float* ub   = carve((size_t)64 * S * S);
    float* fyr = carve(64 * S * MODES); float* fyi = carve(64 * S * MODES);
    float* oyr = carve(64 * S * MODES); float* oyi = carve(64 * S * MODES);
    float* fxr = carve(64 * S * MODES); float* fxi = carve(64 * S * MODES);
    float* oxr = carve(64 * S * MODES); float* oxi = carve(64 * S * MODES);
    float* t1r = carve(64 * M2 * S);    float* t1i = carve(64 * M2 * S);
    float* u2r = carve(64 * M2 * MODES);float* u2i = carve(64 * M2 * MODES);
    float* Ftr = carve(64 * M2 * MODES);float* Fti = carve(64 * M2 * MODES);
    float* Fr  = carve(64 * NMODEP);    float* Fi  = carve(64 * NMODEP);
    float* uo  = carve((size_t)64 * G);

    const int T = 256;
    k_pre<<<(G + T - 1) / T, T, 0, stream>>>(x, coords, pre_w1, pre_b1, pre_w2, pre_b2, u_pre);
    k_iphi<<<G, 128, 0, stream>>>(coords, ifc0w, ifc0b, ifncw, ifncb,
                                  ifc1w, ifc1b, ifc2w, ifc2b, ifc3w, ifc3b,
                                  ifc4w, ifc4b, xphi);
    k_basis<<<(G + T - 1) / T, T, 0, stream>>>(xphi, e1r, e1i, e2r, e2i);

    k_pin_gemm<<<dim3(M2, KCHUNK), 256, 0, stream>>>(u_pre, e1r, e1i, e2r, e2i, partial);
    {
        int nred = 2 * 64 * M2 * MODES;
        k_reduce_pin<<<(nred + T - 1) / T, T, 0, stream>>>(partial, uftr, ufti);
    }
    {
        int n1 = 64 * M2 * MODES;
        k_specmix<<<(n1 + T - 1) / T, T, 0, stream>>>(uftr, ufti, pin_w1_re, pin_w1_im,
                                                      pin_w2_re, pin_w2_im, mxr, mxi);
    }
    {
        int ng1 = 64 * S * MODES;
        k_grid1<<<(ng1 + T - 1) / T, T, 0, stream>>>(mxr, mxi, g1r, g1i);
        int ng2 = 64 * S * S;
        k_grid2<<<(ng2 + T - 1) / T, T, 0, stream>>>(g1r, g1i, ua);
    }

    float* src = ua; float* dst = ub;
    const int NF = 64 * S * MODES;
    const int NG = 64 * S * S;
    for (int l = 0; l < 4; ++l) {
        const float* wyr = spec_wy_re + (size_t)l * 64 * 64 * MODES;
        const float* wyi = spec_wy_im + (size_t)l * 64 * 64 * MODES;
        const float* wxr = spec_wx_re + (size_t)l * 64 * 64 * MODES;
        const float* wxi = spec_wx_im + (size_t)l * 64 * 64 * MODES;
        k_ffty<<<(NF + T - 1) / T, T, 0, stream>>>(src, fyr, fyi);
        k_mixy<<<(NF + T - 1) / T, T, 0, stream>>>(fyr, fyi, wyr, wyi, oyr, oyi);
        k_fftx<<<(NF + T - 1) / T, T, 0, stream>>>(src, fxr, fxi);
        k_mixx<<<(NF + T - 1) / T, T, 0, stream>>>(fxr, fxi, wxr, wxi, oxr, oxi);
        k_combine<<<(NG + T - 1) / T, T, 0, stream>>>(src, oxr, oxi, oyr, oyi, dst);
        float* tmp = src; src = dst; dst = tmp;
    }

    {
        int np1 = 64 * M2 * S;
        k_po1<<<(np1 + T - 1) / T, T, 0, stream>>>(src, t1r, t1i);
        int np2 = 64 * M2 * MODES;
        k_po2<<<(np2 + T - 1) / T, T, 0, stream>>>(t1r, t1i, u2r, u2i);
        k_specmix<<<(np2 + T - 1) / T, T, 0, stream>>>(u2r, u2i, pout_w1_re, pout_w1_im,
                                                       pout_w2_re, pout_w2_im, Ftr, Fti);
        int npk = 64 * NMODEP;
        k_fpack<<<(npk + T - 1) / T, T, 0, stream>>>(Ftr, Fti, Fr, Fi);
    }
    k_pout_gemm<<<G / 32, 256, 0, stream>>>(Fr, Fi, e1r, e1i, e2r, e2i, uo);
    k_head<<<(G + T - 1) / T, T, 0, stream>>>(uo, head_w1, head_b1, head_w2, head_b2,
                                              (float*)d_out);
}
